// MultiHeadAttention_47399259079145
// MI455X (gfx1250) — compile-verified, matrix operand order checked by reading, not run on hardware
//
#include <hip/hip_runtime.h>
#include <hip/hip_bf16.h>
#include <math.h>
#include <stdint.h>

// ---------------- problem constants ----------------
#define B_   4
#define T_   12
#define P_   196
#define D_   512
#define H_   8
#define DH_  64
#define PP   224              // P padded to multiple of 32 for WMMA K-loops
#define MTOK (B_*T_*P_)       // 9408 tokens (rows)
#define KTD  (T_*D_)          // 6144 contraction dim for temporal GEMM

typedef __attribute__((ext_vector_type(16))) _Float16 v16h;
typedef __attribute__((ext_vector_type(8)))  float    v8f;
typedef __attribute__((ext_vector_type(4)))  int      v4i;

#define WMMA_F16(Af, Bf, Cacc) \
  __builtin_amdgcn_wmma_f32_16x16x32_f16(false, (Af), false, (Bf), (short)0, (Cacc), false, false)

// CDNA5 async global->LDS path (ASYNCcnt-tracked), guarded so compile never breaks.
#if defined(__has_builtin)
#  if __has_builtin(__builtin_amdgcn_global_load_async_to_lds_b128) && \
      __has_builtin(__builtin_amdgcn_s_wait_asynccnt)
#    define USE_ASYNC_LDS 1
#  endif
#endif
#ifndef USE_ASYNC_LDS
#  define USE_ASYNC_LDS 0
#endif

__device__ __forceinline__ void cp16_g2l(_Float16* ldst, const _Float16* gsrc) {
#if USE_ASYNC_LDS
  // Builtin signature: (global int4* src, local int4* dst, imm offset, imm cpol).
  // A generic LDS pointer keeps the LDS offset in its low 32 bits (ISA 10.2).
  __builtin_amdgcn_global_load_async_to_lds_b128(
      (__attribute__((address_space(1))) v4i*)(uintptr_t)gsrc,
      (__attribute__((address_space(3))) v4i*)(uint32_t)(uintptr_t)ldst,
      0, 0);
#else
  *(float4*)ldst = *(const float4*)gsrc;
#endif
}

__device__ __forceinline__ void async_join() {
#if USE_ASYNC_LDS
  __builtin_amdgcn_s_wait_asynccnt(0);
#endif
}

__device__ __forceinline__ float gelu_f(float x) {
  return 0.5f * x * (1.0f + erff(x * 0.7071067811865475f));
}

// Load a 16x32 f16 WMMA A/B fragment from a row-major [rows x ld] matrix
// (B operand stored N x K). Per ISA 7.12.2: lanes 0-15 hold
// K = [k0..k0+7] U [k0+16..k0+23]; lanes 16-31 hold K = [k0+8..k0+15] U [k0+24..k0+31].
__device__ __forceinline__ v16h load_frag(const _Float16* base, int row, int ld,
                                          int k0, int lane) {
  const _Float16* p = base + (size_t)row * ld + k0 + ((lane & 16) >> 1);
  union { v16h v; float4 q[2]; } u;
  u.q[0] = *(const float4*)(p);
  u.q[1] = *(const float4*)(p + 16);
  return u.v;
}

// ---------------- WMMA GEMM: C = A * B^T (+bias)(+epilogue) ----------------
// A: [M x K] f16 row-major.  Bm: [N x K] f16 row-major.
// Block = 8 waves (2x4), block tile 64M x 128N, each wave 32x32 (2x2 frags,
// 4 independent accumulators -> no WMMA RAW hazard nops).
// K-panels double-buffered in LDS via async global->LDS copies.
// EPI 0: Cf = acc + bias ; EPI 1: Cf = res + gelu(acc+bias) ; EPI 2: Ch = f16 gelu(acc+bias)
template <int EPI>
__global__ __launch_bounds__(256)
void gemm16(const _Float16* __restrict__ A, const _Float16* __restrict__ Bm,
            int M, int N, int K,
            const float* __restrict__ bias, const float* __restrict__ res,
            float* __restrict__ Cf, _Float16* __restrict__ Ch) {
  __shared__ __align__(16) _Float16 At[2][64 * 32];     // 2 x 4 KB
  __shared__ __align__(16) _Float16 Bt[2][128 * 32];    // 2 x 8 KB
  const int tid = threadIdx.x, lane = tid & 31, wave = tid >> 5;
  const int wm = wave >> 2, wn = wave & 3;
  const int m0 = blockIdx.y * 64;
  const int n0 = blockIdx.x * 128;

  const int crow = tid >> 2;              // 0..63
  const int cseg = (tid & 3) * 8;         // halves within 64B row chunk
  auto copy_tiles = [&](int buf, int k0) {
    cp16_g2l(&At[buf][crow * 32 + cseg],        A  + (size_t)(m0 + crow) * K + k0 + cseg);
    cp16_g2l(&Bt[buf][crow * 32 + cseg],        Bm + (size_t)(n0 + crow) * K + k0 + cseg);
    cp16_g2l(&Bt[buf][(64 + crow) * 32 + cseg], Bm + (size_t)(n0 + 64 + crow) * K + k0 + cseg);
  };

  const int rA0 = wm * 32 + (lane & 15), rA1 = rA0 + 16;
  const int rB0 = wn * 32 + (lane & 15), rB1 = rB0 + 16;
  v8f a00 = {}, a01 = {}, a10 = {}, a11 = {};

  copy_tiles(0, 0);
  const int nk = K >> 5;
  for (int kt = 0; kt < nk; ++kt) {
    const int buf = kt & 1;
    async_join();
    __syncthreads();                        // panel ready + prev compute done
    if (kt + 1 < nk) copy_tiles(buf ^ 1, (kt + 1) << 5);
    v16h fa0 = load_frag(At[buf], rA0, 32, 0, lane);
    v16h fa1 = load_frag(At[buf], rA1, 32, 0, lane);
    v16h fb0 = load_frag(Bt[buf], rB0, 32, 0, lane);
    v16h fb1 = load_frag(Bt[buf], rB1, 32, 0, lane);
    a00 = WMMA_F16(fa0, fb0, a00);
    a01 = WMMA_F16(fa0, fb1, a01);
    a10 = WMMA_F16(fa1, fb0, a10);
    a11 = WMMA_F16(fa1, fb1, a11);
  }

  const int nn = n0 + wn * 32 + (lane & 15);
  const int mb = m0 + wm * 32 + ((lane & 16) >> 1);
  const float bv0 = bias ? bias[nn] : 0.0f;
  const float bv1 = bias ? bias[nn + 16] : 0.0f;
  auto store_acc = [&](v8f acc, int mbase, int n, float bv) {
#pragma unroll
    for (int r = 0; r < 8; ++r) {
      const size_t o = (size_t)(mbase + r) * N + n;
      const float v = acc[r] + bv;
      if (EPI == 0)      Cf[o] = v;
      else if (EPI == 1) Cf[o] = res[o] + gelu_f(v);
      else               Ch[o] = (_Float16)gelu_f(v);
    }
  };
  store_acc(a00, mb,      nn,      bv0);
  store_acc(a01, mb,      nn + 16, bv1);
  store_acc(a10, mb + 16, nn,      bv0);
  store_acc(a11, mb + 16, nn + 16, bv1);
}

// ---------------- elementwise / packing kernels ----------------
__global__ void cast_f32_f16(const float* __restrict__ s, _Float16* __restrict__ d, int n) {
  int i = blockIdx.x * blockDim.x + threadIdx.x;
  if (i < n) d[i] = (_Float16)s[i];
}

// Wt [T,D,D] (u,e,d) -> WtH [e][(u*D + d)] f16  (B operand, N x K layout)
__global__ void pack_wt(const float* __restrict__ Wt, _Float16* __restrict__ o) {
  int i = blockIdx.x * blockDim.x + threadIdx.x;
  if (i >= D_ * KTD) return;
  int d = i % D_, u = (i / D_) % T_, e = i / (D_ * T_);
  o[(size_t)e * KTD + u * D_ + d] = (_Float16)Wt[((size_t)u * D_ + e) * D_ + d];
}

__global__ void btsum_k(const float* __restrict__ bt, float* __restrict__ o) {
  int e = blockIdx.x * blockDim.x + threadIdx.x;
  if (e >= D_) return;
  float s = 0.f;
  for (int u = 0; u < T_; ++u) s += bt[u * D_ + e];
  o[e] = s;
}

// Row LayerNorm (alpha*(x-mu)/(std_ddof1+eps)+beta) -> f16 rows of D
__global__ __launch_bounds__(128)
void row_norm_f16(const float* __restrict__ X, const float* __restrict__ alpha,
                  const float* __restrict__ beta, _Float16* __restrict__ Y) {
  const int row = blockIdx.x, tid = threadIdx.x, wave = tid >> 5;
  const float* xr = X + (size_t)row * D_;
  float v[4], s = 0.f, q = 0.f;
#pragma unroll
  for (int i = 0; i < 4; ++i) { float x = xr[i * 128 + tid]; v[i] = x; s += x; q += x * x; }
  for (int off = 16; off; off >>= 1) { s += __shfl_xor(s, off, 32); q += __shfl_xor(q, off, 32); }
  __shared__ float rs[4], rq[4];
  if ((tid & 31) == 0) { rs[wave] = s; rq[wave] = q; }
  __syncthreads();
  s = rs[0] + rs[1] + rs[2] + rs[3];
  q = rq[0] + rq[1] + rq[2] + rq[3];
  const float mu = s * (1.0f / D_);
  const float var = (q - (float)D_ * mu * mu) * (1.0f / (D_ - 1));
  const float rinv = 1.0f / (sqrtf(fmaxf(var, 0.f)) + 1e-6f);
#pragma unroll
  for (int i = 0; i < 4; ++i) {
    int d = i * 128 + tid;
    Y[(size_t)row * D_ + d] = (_Float16)(alpha[d] * (v[i] - mu) * rinv + beta[d]);
  }
}

// Pack QKV f32 [ (b,t,p) x (h,dh) ] into padded per-head f16 tiles.
// tr==0: dst[bth][p(224)][dh]; tr==1: dst[bth][dh][p(224)] (B operand for AV)
__global__ void pack_qkv(const float* __restrict__ S, _Float16* __restrict__ dst, int tr) {
  int i = blockIdx.x * blockDim.x + threadIdx.x;
  const int total = B_ * T_ * H_ * PP * DH_;
  if (i >= total) return;
  int dh = i % DH_, p = (i / DH_) % PP, bth = i / (DH_ * PP);
  int h = bth % H_, bt = bth / H_;
  float v = (p < P_) ? S[((size_t)bt * P_ + p) * D_ + h * DH_ + dh] : 0.0f;
  if (!tr) dst[i] = (_Float16)v;
  else     dst[((size_t)bth * DH_ + dh) * PP + p] = (_Float16)v;
}

// ---------------- attention: one block per (b,t,u,h) ----------------
__global__ __launch_bounds__(256)
void attn_kernel(const _Float16* __restrict__ Qp, const _Float16* __restrict__ Kp,
                 const _Float16* __restrict__ Vt, float* __restrict__ av) {
  __shared__ __align__(16) _Float16 Sh[PP * PP];   // ~100 KB of the 320 KB WGP LDS
  const int h = blockIdx.x;
  const int t = blockIdx.y / T_, u = blockIdx.y % T_;
  const int b = blockIdx.z;
  const int tid = threadIdx.x, lane = tid & 31, wave = tid >> 5;

  const _Float16* qB = Qp + (size_t)((b * T_ + t) * H_ + h) * PP * DH_;
  const _Float16* kB = Kp + (size_t)((b * T_ + u) * H_ + h) * PP * DH_;
  const _Float16* vB = Vt + (size_t)((b * T_ + u) * H_ + h) * DH_ * PP;

  // Stage 1: S = (Q K^T)/8 into LDS. 7x7 macro-tiles of 32x32, 2x2 register blocking.
  for (int idx = wave; idx < 49; idx += 8) {
    const int rt = (idx / 7) * 32, ct = (idx % 7) * 32;
    const int ra0 = rt + (lane & 15), ra1 = ra0 + 16;
    const int rb0 = ct + (lane & 15), rb1 = rb0 + 16;
    v8f s00 = {}, s01 = {}, s10 = {}, s11 = {};
    for (int k0 = 0; k0 < DH_; k0 += 32) {
      v16h qa0 = load_frag(qB, ra0, DH_, k0, lane);
      v16h qa1 = load_frag(qB, ra1, DH_, k0, lane);
      v16h kb0 = load_frag(kB, rb0, DH_, k0, lane);
      v16h kb1 = load_frag(kB, rb1, DH_, k0, lane);
      s00 = WMMA_F16(qa0, kb0, s00);
      s01 = WMMA_F16(qa0, kb1, s01);
      s10 = WMMA_F16(qa1, kb0, s10);
      s11 = WMMA_F16(qa1, kb1, s11);
    }
    const int mb = rt + ((lane & 16) >> 1);
    const int nn = ct + (lane & 15);
#pragma unroll
    for (int r = 0; r < 8; ++r) {
      Sh[(mb + r) * PP + nn]           = (_Float16)(s00[r] * 0.125f);
      Sh[(mb + r) * PP + nn + 16]      = (_Float16)(s01[r] * 0.125f);
      Sh[(mb + r + 16) * PP + nn]      = (_Float16)(s10[r] * 0.125f);
      Sh[(mb + r + 16) * PP + nn + 16] = (_Float16)(s11[r] * 0.125f);
    }
  }
  __syncthreads();

  // Stage 2: softmax over valid 196 cols; padded cols forced to exactly 0.
  for (int row = wave; row < P_; row += 8) {
    float vals[7], mx = -3.0e38f;
#pragma unroll
    for (int i = 0; i < 7; ++i) {
      int c = i * 32 + lane;
      float v = (c < P_) ? (float)Sh[row * PP + c] : -3.0e38f;
      vals[i] = v; mx = fmaxf(mx, v);
    }
    for (int off = 16; off; off >>= 1) mx = fmaxf(mx, __shfl_xor(mx, off, 32));
    float s = 0.f;
#pragma unroll
    for (int i = 0; i < 7; ++i) {
      int c = i * 32 + lane;
      float e = (c < P_) ? __expf(vals[i] - mx) : 0.0f;
      vals[i] = e; s += e;
    }
    for (int off = 16; off; off >>= 1) s += __shfl_xor(s, off, 32);
    const float inv = 1.0f / s;
#pragma unroll
    for (int i = 0; i < 7; ++i)
      Sh[row * PP + i * 32 + lane] = (_Float16)(vals[i] * inv);
  }
  __syncthreads();

  // Stage 3: AV. 7x2 macro-tiles (rows 224 x cols 64), 2x2 blocking, K = 224 from LDS.
  float* avb = av + (size_t)(((b * T_ + t) * T_ + u)) * P_ * D_ + h * DH_;
  for (int idx = wave; idx < 14; idx += 8) {
    const int rt = (idx >> 1) * 32, cn = (idx & 1) * 32;
    const int ra0 = rt + (lane & 15), ra1 = ra0 + 16;
    const int rb0 = cn + (lane & 15), rb1 = rb0 + 16;
    v8f s00 = {}, s01 = {}, s10 = {}, s11 = {};
    for (int k0 = 0; k0 < PP; k0 += 32) {
      v16h pa0 = load_frag(Sh, ra0, PP, k0, lane);
      v16h pa1 = load_frag(Sh, ra1, PP, k0, lane);
      v16h vb0 = load_frag(vB, rb0, PP, k0, lane);
      v16h vb1 = load_frag(vB, rb1, PP, k0, lane);
      s00 = WMMA_F16(pa0, vb0, s00);
      s01 = WMMA_F16(pa0, vb1, s01);
      s10 = WMMA_F16(pa1, vb0, s10);
      s11 = WMMA_F16(pa1, vb1, s11);
    }
    const int mb = rt + ((lane & 16) >> 1);
    const int nn = cn + (lane & 15);
#pragma unroll
    for (int r = 0; r < 8; ++r) {
      const int p0 = mb + r, p1 = mb + r + 16;
      if (p0 < P_) {
        avb[(size_t)p0 * D_ + nn]      = s00[r];
        avb[(size_t)p0 * D_ + nn + 16] = s01[r];
      }
      if (p1 < P_) {
        avb[(size_t)p1 * D_ + nn]      = s10[r];
        avb[(size_t)p1 * D_ + nn + 16] = s11[r];
      }
    }
  }
}

// Norm(av) + pos, scale 1/T, scatter f16 into A layout [(b,t,p)][(u*D+d)].
__global__ __launch_bounds__(128)
void attn_norm_pos_pack(const float* __restrict__ av, const float* __restrict__ alpha,
                        const float* __restrict__ beta, const float* __restrict__ pos,
                        _Float16* __restrict__ x2ph) {
  const int row = blockIdx.x;                 // ((b*T+t)*T+u)*P + p
  const int p = row % P_;
  const int u = (row / P_) % T_;
  const int t = (row / (P_ * T_)) % T_;
  const int b = row / (P_ * T_ * T_);
  const int tid = threadIdx.x, wave = tid >> 5;
  const float* xr = av + (size_t)row * D_;
  float v[4], s = 0.f, q = 0.f;
#pragma unroll
  for (int i = 0; i < 4; ++i) { float x = xr[i * 128 + tid]; v[i] = x; s += x; q += x * x; }
  for (int off = 16; off; off >>= 1) { s += __shfl_xor(s, off, 32); q += __shfl_xor(q, off, 32); }
  __shared__ float rs[4], rq[4];
  if ((tid & 31) == 0) { rs[wave] = s; rq[wave] = q; }
  __syncthreads();
  s = rs[0] + rs[1] + rs[2] + rs[3];
  q = rq[0] + rq[1] + rq[2] + rq[3];
  const float mu = s * (1.0f / D_);
  const float var = (q - (float)D_ * mu * mu) * (1.0f / (D_ - 1));
  const float rinv = 1.0f / (sqrtf(fmaxf(var, 0.f)) + 1e-6f);
  const float* pr = pos + (size_t)((t * T_ + u) * P_ + p) * D_;
  _Float16* dst = x2ph + (size_t)((b * T_ + t) * P_ + p) * KTD + u * D_;
#pragma unroll
  for (int i = 0; i < 4; ++i) {
    int d = i * 128 + tid;
    float val = alpha[d] * (v[i] - mu) * rinv + beta[d] + pr[d];
    dst[d] = (_Float16)(val * (1.0f / T_));
  }
}

// ---------------- host orchestration ----------------
extern "C" void kernel_launch(void* const* d_in, const int* in_sizes, int n_in,
                              void* d_out, int out_size, void* d_ws, size_t ws_size,
                              hipStream_t stream) {
  const float* x    = (const float*)d_in[0];
  const float* Wq   = (const float*)d_in[1];
  const float* bq   = (const float*)d_in[2];
  const float* Wk   = (const float*)d_in[3];
  const float* bk   = (const float*)d_in[4];
  const float* Wv   = (const float*)d_in[5];
  const float* bv   = (const float*)d_in[6];
  const float* in_a = (const float*)d_in[7];
  const float* in_b = (const float*)d_in[8];
  const float* at_a = (const float*)d_in[9];
  const float* at_b = (const float*)d_in[10];
  const float* ot_a = (const float*)d_in[11];
  const float* ot_b = (const float*)d_in[12];
  const float* Wt   = (const float*)d_in[13];
  const float* bt   = (const float*)d_in[14];
  const float* pos  = (const float*)d_in[15];
  const float* W1   = (const float*)d_in[16];
  const float* b1   = (const float*)d_in[17];
  const float* W2   = (const float*)d_in[18];
  const float* b2   = (const float*)d_in[19];
  float* out = (float*)d_out;
  (void)in_sizes; (void)n_in; (void)out_size; (void)ws_size;

  char* wp = (char*)d_ws;
  auto alloc = [&](size_t bytes) -> void* {
    void* p = (void*)wp;
    wp += (bytes + 255) & ~(size_t)255;
    return p;
  };

  _Float16* x2h   = (_Float16*)alloc((size_t)MTOK * D_ * 2);
  _Float16* WqH   = (_Float16*)alloc((size_t)D_ * D_ * 2);
  _Float16* WkH   = (_Float16*)alloc((size_t)D_ * D_ * 2);
  _Float16* WvH   = (_Float16*)alloc((size_t)D_ * D_ * 2);
  _Float16* W1H   = (_Float16*)alloc((size_t)2 * D_ * D_ * 2);
  _Float16* W2H   = (_Float16*)alloc((size_t)2 * D_ * D_ * 2);
  _Float16* WtH   = (_Float16*)alloc((size_t)D_ * KTD * 2);
  float*    btsum = (float*)alloc((size_t)D_ * 4);
  float*    Qf    = (float*)alloc((size_t)MTOK * D_ * 4);
  float*    Kf    = (float*)alloc((size_t)MTOK * D_ * 4);
  float*    Vf    = (float*)alloc((size_t)MTOK * D_ * 4);
  _Float16* Qp    = (_Float16*)alloc((size_t)B_ * T_ * H_ * PP * DH_ * 2);
  _Float16* Kp    = (_Float16*)alloc((size_t)B_ * T_ * H_ * PP * DH_ * 2);
  _Float16* Vtb   = (_Float16*)alloc((size_t)B_ * T_ * H_ * PP * DH_ * 2);
  float*    av    = (float*)alloc((size_t)B_ * T_ * T_ * P_ * D_ * 4);
  _Float16* x2ph  = (_Float16*)alloc((size_t)MTOK * KTD * 2);
  // Reused regions (producers/consumers strictly ordered on the stream):
  float*    xmid  = Qf;
  _Float16* x2bh  = (_Float16*)Kf;
  _Float16* h1h   = (_Float16*)Vf;            // 9408*1024 f16 fits exactly

  // 1) weight casts / packs
  cast_f32_f16<<<(D_ * D_ + 255) / 256, 256, 0, stream>>>(Wq, WqH, D_ * D_);
  cast_f32_f16<<<(D_ * D_ + 255) / 256, 256, 0, stream>>>(Wk, WkH, D_ * D_);
  cast_f32_f16<<<(D_ * D_ + 255) / 256, 256, 0, stream>>>(Wv, WvH, D_ * D_);
  cast_f32_f16<<<(2 * D_ * D_ + 255) / 256, 256, 0, stream>>>(W1, W1H, 2 * D_ * D_);
  cast_f32_f16<<<(2 * D_ * D_ + 255) / 256, 256, 0, stream>>>(W2, W2H, 2 * D_ * D_);
  pack_wt<<<(D_ * KTD + 255) / 256, 256, 0, stream>>>(Wt, WtH);
  btsum_k<<<2, 256, 0, stream>>>(bt, btsum);

  // 2) input norm -> f16
  row_norm_f16<<<MTOK, 128, 0, stream>>>(x, in_a, in_b, x2h);

  // 3) QKV projections (WMMA, LDS double-buffered panels)
  dim3 gD(D_ / 128, MTOK / 64);
  gemm16<0><<<gD, 256, 0, stream>>>(x2h, WqH, MTOK, D_, D_, bq, nullptr, Qf, nullptr);
  gemm16<0><<<gD, 256, 0, stream>>>(x2h, WkH, MTOK, D_, D_, bk, nullptr, Kf, nullptr);
  gemm16<0><<<gD, 256, 0, stream>>>(x2h, WvH, MTOK, D_, D_, bv, nullptr, Vf, nullptr);

  // 4) pack per-head padded tiles
  const int npk = B_ * T_ * H_ * PP * DH_;
  pack_qkv<<<(npk + 255) / 256, 256, 0, stream>>>(Qf, Qp, 0);
  pack_qkv<<<(npk + 255) / 256, 256, 0, stream>>>(Kf, Kp, 0);
  pack_qkv<<<(npk + 255) / 256, 256, 0, stream>>>(Vf, Vtb, 1);

  // 5) attention (WMMA + LDS softmax), one block per (b,t,u,h)
  attn_kernel<<<dim3(H_, T_ * T_, B_), 256, 0, stream>>>(Qp, Kp, Vtb, av);

  // 6) norm + pos + 1/T, packed for the temporal contraction
  attn_norm_pos_pack<<<B_ * T_ * T_ * P_, 128, 0, stream>>>(av, at_a, at_b, pos, x2ph);

  // 7) temporal GEMM (K = 6144) fused residual + exact GELU: xmid = x + gelu(tc)
  gemm16<1><<<gD, 256, 0, stream>>>(x2ph, WtH, MTOK, D_, KTD, btsum, x, xmid, nullptr);

  // 8) FFN
  row_norm_f16<<<MTOK, 128, 0, stream>>>(xmid, ot_a, ot_b, x2bh);
  gemm16<2><<<dim3(2 * D_ / 128, MTOK / 64), 256, 0, stream>>>(
      x2bh, W1H, MTOK, 2 * D_, D_, b1, nullptr, nullptr, h1h);
  gemm16<1><<<gD, 256, 0, stream>>>(h1h, W2H, MTOK, D_, 2 * D_, b2, xmid, out, nullptr);
}
